// MultiheadAttention_88587995447535
// MI455X (gfx1250) — matrix operand order repaired and checked by reading, not compiled, not run on hardware
//
#include <hip/hip_runtime.h>
#include <hip/hip_bf16.h>

#define B_ 4
#define S_ 2048
#define E_ 1024
#define H_ 16
#define D_ 64
#define M_ (B_ * S_) // 8192

#define BLD 40                 // padded LDS leading dim (elements) for 64x32 B tiles
#define BTILE (64 * BLD)       // elements per LDS buffer

static __device__ __forceinline__ float inv_scale() { return 0.125f; } // 1/sqrt(64)

typedef __bf16 bf16_t;
typedef __attribute__((ext_vector_type(16))) __bf16 v16bf;
typedef __attribute__((ext_vector_type(8)))  __bf16 v8bf;
typedef __attribute__((ext_vector_type(8)))  float  v8f;

union V16BF { v16bf v; v8bf h[2]; __bf16 e[16]; };

// ---- WMMA fragment loaders (wave32; this code's contiguous k = 16*(lane/16) + s order in both A and B) ----

// A operand: 16x32 bf16 tile from row-major source (K contiguous), ld in elements.
__device__ __forceinline__ v16bf load_a_bf16(const bf16_t* __restrict__ base, int lda, int lane) {
  const int m  = lane & 15;
  const int kb = (lane < 16) ? 0 : 16;
  const bf16_t* p = base + (size_t)m * lda + kb;
  V16BF u;
  u.h[0] = *(const v8bf*)(p);        // K = kb .. kb+7   -> VGPRs 0-3
  u.h[1] = *(const v8bf*)(p + 8);    // K = kb+8..kb+15  -> VGPRs 4-7
  return u.v;
}

// B operand from global: 32x16 bf16 tile; source row-major [n][k] (rows = columns of B).
__device__ __forceinline__ v16bf load_b_bf16(const bf16_t* __restrict__ base, int ldb, int lane) {
  const int n  = lane & 15;
  const int kb = (lane < 16) ? 0 : 16;
  return *(const v16bf*)(base + (size_t)n * ldb + kb);  // 16 contiguous K values
}

// B operand from LDS tile with padded ld (16B-aligned halves to tolerate BLD=40).
__device__ __forceinline__ v16bf load_b_lds(const bf16_t* base, int lane) {
  const int n  = lane & 15;
  const int kb = (lane < 16) ? 0 : 16;
  const bf16_t* p = base + n * BLD + kb;
  V16BF u;
  u.h[0] = *(const v8bf*)(p);
  u.h[1] = *(const v8bf*)(p + 8);
  return u.v;
}

// A operand from fp32 source (attention weights), NT loads + convert to bf16.
__device__ __forceinline__ v16bf load_a_f32(const float* __restrict__ base, int lda, int lane) {
  const int m  = lane & 15;
  const int kb = (lane < 16) ? 0 : 16;
  const float* p = base + (size_t)m * lda + kb;
  v8f f0 = __builtin_nontemporal_load((const v8f*)(p));
  v8f f1 = __builtin_nontemporal_load((const v8f*)(p + 8));
  V16BF u;
#pragma unroll
  for (int i = 0; i < 8; ++i) { u.e[i] = (__bf16)f0[i]; u.e[8 + i] = (__bf16)f1[i]; }
  return u.v;
}

__device__ __forceinline__ v8f wmma_bf16(v16bf a, v16bf b, v8f c) {
  return __builtin_amdgcn_wmma_f32_16x16x32_bf16(false, a, false, b, (short)0, c, false, false);
}

// ---- CDNA5 async global->LDS copy of one 16B chunk (ASYNCcnt-tracked DMA) ----
__device__ __forceinline__ void async_copy_b128(const bf16_t* lds_dst, const bf16_t* gsrc) {
  const unsigned lds_off = (unsigned)(unsigned long long)(const void*)lds_dst;
  asm volatile("global_load_async_to_lds_b128 %0, %1, off"
               :: "v"(lds_off), "v"((unsigned long long)(const void*)gsrc)
               : "memory");
}

// ---- fp32 -> bf16 convert ----
__global__ void mha_cvt_bf16(const float* __restrict__ in, bf16_t* __restrict__ out, int n) {
  int i = blockIdx.x * blockDim.x + threadIdx.x;
  if (i < n) out[i] = (bf16_t)in[i];
}

// ---- fused QKV projection: Y = x @ W.T + b ; grid.z selects (wq->Q, wk->K, wv->Vt) ----
// Block = 4 waves = 64 rows x 64 cols; B tile (64x32) staged in LDS via async DMA,
// double buffered; A fragment register double-buffered.
__global__ void mha_qkv_gemm(const bf16_t* __restrict__ xb,
                             const bf16_t* __restrict__ wqb,
                             const bf16_t* __restrict__ wkb,
                             const bf16_t* __restrict__ wvb,
                             const float* __restrict__ bq,
                             const float* __restrict__ bk,
                             const float* __restrict__ bv,
                             bf16_t* __restrict__ Q,
                             bf16_t* __restrict__ K,
                             bf16_t* __restrict__ Vt) {
  __shared__ __align__(32) bf16_t Btile[2][BTILE];
  const int tid  = threadIdx.x;
  const int lane = tid & 31;
  const int wave = tid >> 5;
  const int m0 = (blockIdx.x * 4 + wave) * 16;
  const int n0 = blockIdx.y * 64;
  const int which = blockIdx.z;
  const bf16_t* W   = (which == 0) ? wqb : (which == 1) ? wkb : wvb;
  const float* bias = (which == 0) ? bq  : (which == 1) ? bk  : bv;
  const bf16_t* Wbase = W + (size_t)n0 * E_;

  const int srow = tid >> 2;   // 0..31
  const int sc   = tid & 3;    // 16B chunk 0..3
  auto stage = [&](int k, int buf) {
#pragma unroll
    for (int it = 0; it < 2; ++it) {
      const int r = srow + it * 32;
      async_copy_b128(&Btile[buf][r * BLD + sc * 8], Wbase + (size_t)r * E_ + k + sc * 8);
    }
  };

  stage(0, 0);
  v16bf a_cur = load_a_bf16(xb + (size_t)m0 * E_, E_, lane);

  v8f zero = {};
  v8f acc[4] = {zero, zero, zero, zero};
  for (int k = 0; k < E_; k += 32) {
    const int buf = (k >> 5) & 1;
    const bool more = (k + 32 < E_);
    if (more) {
      stage(k + 32, buf ^ 1);
      asm volatile("s_wait_asynccnt 0x2" ::: "memory");  // current tile resident
    } else {
      asm volatile("s_wait_asynccnt 0x0" ::: "memory");
    }
    __syncthreads();

    v16bf a_nxt = a_cur;
    if (more) a_nxt = load_a_bf16(xb + (size_t)m0 * E_ + k + 32, E_, lane);

    const bf16_t* Bl = &Btile[buf][0];
#pragma unroll
    for (int nt = 0; nt < 4; ++nt) {
      v16bf b = load_b_lds(Bl + nt * 16 * BLD, lane);
      acc[nt] = wmma_bf16(a_cur, b, acc[nt]);
    }
    a_cur = a_nxt;
    __syncthreads();  // all waves done reading buf before it is restaged
  }

  const int mb = (lane < 16) ? 0 : 8;
  const int nn = lane & 15;
#pragma unroll
  for (int nt = 0; nt < 4; ++nt) {
    const int n = n0 + nt * 16 + nn;
    const int h = n >> 6;      // n / D
    const int d = n & 63;      // n % D
    const float bval = bias[n];
#pragma unroll
    for (int r = 0; r < 8; ++r) {
      const int m = m0 + mb + r;
      const int b = m >> 11;   // m / S
      const int s = m & 2047;  // m % S
      const float y = acc[nt][r] + bval;
      if (which == 2) {
        Vt[((size_t)(b * H_ + h) * D_ + d) * S_ + s] = (bf16_t)y;   // V transposed [b,h,d,s]
      } else {
        bf16_t* dst = (which == 0) ? Q : K;
        dst[((size_t)(b * H_ + h) * S_ + s) * D_ + d] = (bf16_t)y;  // [b,h,s,d]
      }
    }
  }
}

// ---- raw scores: attn = Q @ K^T / sqrt(D), causal mask, NT stores (1.07GB stream) ----
__global__ void mha_scores(const bf16_t* __restrict__ Q,
                           const bf16_t* __restrict__ K,
                           float* __restrict__ attn,
                           const int* __restrict__ dmask) {
  const int lane = threadIdx.x & 31;
  const int wave = threadIdx.x >> 5;
  const int bh = blockIdx.y;
  const int q0 = (blockIdx.x * 4 + wave) * 16;
  const bool causal = (*dmask != 0);

  const bf16_t* Qbh = Q + (size_t)bh * S_ * D_;
  const bf16_t* Kbh = K + (size_t)bh * S_ * D_;
  float* Abh = attn + (size_t)bh * S_ * S_;

  const v16bf a0 = load_a_bf16(Qbh + (size_t)q0 * D_ + 0,  D_, lane);
  const v16bf a1 = load_a_bf16(Qbh + (size_t)q0 * D_ + 32, D_, lane);

  const int mb = (lane < 16) ? 0 : 8;
  const int nn = lane & 15;

  for (int k0 = 0; k0 < S_; k0 += 16) {
    if (causal && k0 > q0 + 15) {          // fully masked tile: skip WMMA
#pragma unroll
      for (int r = 0; r < 8; ++r)
        __builtin_nontemporal_store(-INFINITY, &Abh[(size_t)(q0 + mb + r) * S_ + k0 + nn]);
      continue;
    }
    v16bf b0 = load_b_bf16(Kbh + (size_t)k0 * D_ + 0,  D_, lane);
    v16bf b1 = load_b_bf16(Kbh + (size_t)k0 * D_ + 32, D_, lane);
    v8f c = {};
    c = wmma_bf16(a0, b0, c);
    c = wmma_bf16(a1, b1, c);
#pragma unroll
    for (int r = 0; r < 8; ++r) {
      const int row = q0 + mb + r;
      const int col = k0 + nn;
      float v = c[r] * inv_scale();
      if (causal && col > row) v = -INFINITY;
      __builtin_nontemporal_store(v, &Abh[(size_t)row * S_ + col]);
    }
  }
}

// ---- row softmax over 2048 columns, one block per row, register-resident, NT R/W ----
__global__ void mha_softmax(float* __restrict__ attn) {
  __shared__ float red[256];
  const int tid = threadIdx.x;
  float* p = attn + (size_t)blockIdx.x * S_;
  float v[8];
  float m = -INFINITY;
#pragma unroll
  for (int j = 0; j < 8; ++j) {
    v[j] = __builtin_nontemporal_load(&p[tid + j * 256]);
    m = fmaxf(m, v[j]);
  }
  red[tid] = m; __syncthreads();
  for (int s = 128; s > 0; s >>= 1) { if (tid < s) red[tid] = fmaxf(red[tid], red[tid + s]); __syncthreads(); }
  const float rowmax = red[0];
  __syncthreads();
  float sum = 0.f;
#pragma unroll
  for (int j = 0; j < 8; ++j) { v[j] = __expf(v[j] - rowmax); sum += v[j]; }
  red[tid] = sum; __syncthreads();
  for (int s = 128; s > 0; s >>= 1) { if (tid < s) red[tid] += red[tid + s]; __syncthreads(); }
  const float inv = 1.0f / red[0];
#pragma unroll
  for (int j = 0; j < 8; ++j)
    __builtin_nontemporal_store(v[j] * inv, &p[tid + j * 256]);
}

// ---- attn_out = attn @ V, causal loop truncation; output bf16 [b,s,h,d] == [B,S,E] ----
__global__ void mha_av(const float* __restrict__ attn,
                       const bf16_t* __restrict__ Vt,
                       bf16_t* __restrict__ ao,
                       const int* __restrict__ dmask) {
  const int lane = threadIdx.x & 31;
  const int wave = threadIdx.x >> 5;
  const int bh = blockIdx.y;
  const int b = bh >> 4;
  const int h = bh & 15;
  const int q0 = (blockIdx.x * 4 + wave) * 16;
  const bool causal = (*dmask != 0);

  const float*  Abh = attn + (size_t)bh * S_ * S_;
  const bf16_t* Vbh = Vt   + (size_t)bh * D_ * S_;

  const int kmax = causal ? (((q0 + 16) + 31) & ~31) : S_;

  v8f zero = {};
  v8f acc[4] = {zero, zero, zero, zero};
  for (int kk = 0; kk < kmax; kk += 32) {
    v16bf a = load_a_f32(Abh + (size_t)q0 * S_ + kk, S_, lane);
#pragma unroll
    for (int nt = 0; nt < 4; ++nt) {
      v16bf bb = load_b_bf16(Vbh + (size_t)(nt * 16) * S_ + kk, S_, lane);
      acc[nt] = wmma_bf16(a, bb, acc[nt]);
    }
  }

  const int mb = (lane < 16) ? 0 : 8;
  const int nn = lane & 15;
#pragma unroll
  for (int nt = 0; nt < 4; ++nt) {
    const int d = nt * 16 + nn;
#pragma unroll
    for (int r = 0; r < 8; ++r) {
      const int q = q0 + mb + r;
      ao[(((size_t)b * S_ + q) * H_ + h) * D_ + d] = (bf16_t)acc[nt][r];
    }
  }
}

// ---- output projection: out = attn_out @ fc_w.T + fc_b (fp32 out), LDS-staged B ----
__global__ void mha_fc(const bf16_t* __restrict__ ao,
                       const bf16_t* __restrict__ fcw,
                       const float* __restrict__ fcb,
                       float* __restrict__ out) {
  __shared__ __align__(32) bf16_t Btile[2][BTILE];
  const int tid  = threadIdx.x;
  const int lane = tid & 31;
  const int wave = tid >> 5;
  const int m0 = (blockIdx.x * 4 + wave) * 16;
  const int n0 = blockIdx.y * 64;
  const bf16_t* Wbase = fcw + (size_t)n0 * E_;

  const int srow = tid >> 2;
  const int sc   = tid & 3;
  auto stage = [&](int k, int buf) {
#pragma unroll
    for (int it = 0; it < 2; ++it) {
      const int r = srow + it * 32;
      async_copy_b128(&Btile[buf][r * BLD + sc * 8], Wbase + (size_t)r * E_ + k + sc * 8);
    }
  };

  stage(0, 0);
  v16bf a_cur = load_a_bf16(ao + (size_t)m0 * E_, E_, lane);

  v8f zero = {};
  v8f acc[4] = {zero, zero, zero, zero};
  for (int k = 0; k < E_; k += 32) {
    const int buf = (k >> 5) & 1;
    const bool more = (k + 32 < E_);
    if (more) {
      stage(k + 32, buf ^ 1);
      asm volatile("s_wait_asynccnt 0x2" ::: "memory");
    } else {
      asm volatile("s_wait_asynccnt 0x0" ::: "memory");
    }
    __syncthreads();

    v16bf a_nxt = a_cur;
    if (more) a_nxt = load_a_bf16(ao + (size_t)m0 * E_ + k + 32, E_, lane);

    const bf16_t* Bl = &Btile[buf][0];
#pragma unroll
    for (int nt = 0; nt < 4; ++nt) {
      v16bf b = load_b_lds(Bl + nt * 16 * BLD, lane);
      acc[nt] = wmma_bf16(a_cur, b, acc[nt]);
    }
    a_cur = a_nxt;
    __syncthreads();
  }

  const int mb = (lane < 16) ? 0 : 8;
  const int nn = lane & 15;
#pragma unroll
  for (int nt = 0; nt < 4; ++nt) {
    const int n = n0 + nt * 16 + nn;
    const float bval = fcb[n];
#pragma unroll
    for (int r = 0; r < 8; ++r)
      out[(size_t)(m0 + mb + r) * E_ + n] = acc[nt][r] + bval;
  }
}

extern "C" void kernel_launch(void* const* d_in, const int* in_sizes, int n_in,
                              void* d_out, int out_size, void* d_ws, size_t ws_size,
                              hipStream_t stream) {
  const float* x    = (const float*)d_in[0];
  const float* wq_w = (const float*)d_in[1];
  const float* wq_b = (const float*)d_in[2];
  const float* wk_w = (const float*)d_in[3];
  const float* wk_b = (const float*)d_in[4];
  const float* wv_w = (const float*)d_in[5];
  const float* wv_b = (const float*)d_in[6];
  const float* fc_w = (const float*)d_in[7];
  const float* fc_b = (const float*)d_in[8];
  const int*   dm   = (const int*)d_in[9];

  float* out  = (float*)d_out;
  float* attn = out + (size_t)B_ * S_ * E_;   // second tuple output

  char* ws = (char*)d_ws;
  size_t off = 0;
  auto carve = [&](size_t elems) -> bf16_t* {
    bf16_t* p = (bf16_t*)(ws + off);
    off += (elems * sizeof(bf16_t) + 255) & ~(size_t)255;
    return p;
  };
  bf16_t* xb   = carve((size_t)M_ * E_);
  bf16_t* wqb  = carve((size_t)E_ * E_);
  bf16_t* wkb  = carve((size_t)E_ * E_);
  bf16_t* wvb  = carve((size_t)E_ * E_);
  bf16_t* fcwb = carve((size_t)E_ * E_);
  bf16_t* Qb   = carve((size_t)M_ * E_);
  bf16_t* Kb   = carve((size_t)M_ * E_);
  bf16_t* Vtb  = carve((size_t)M_ * E_);
  bf16_t* aob  = carve((size_t)M_ * E_);

  const int nx = M_ * E_;
  const int nw = E_ * E_;
  mha_cvt_bf16<<<(nx + 255) / 256, 256, 0, stream>>>(x, xb, nx);
  mha_cvt_bf16<<<(nw + 255) / 256, 256, 0, stream>>>(wq_w, wqb, nw);
  mha_cvt_bf16<<<(nw + 255) / 256, 256, 0, stream>>>(wk_w, wkb, nw);
  mha_cvt_bf16<<<(nw + 255) / 256, 256, 0, stream>>>(wv_w, wvb, nw);
  mha_cvt_bf16<<<(nw + 255) / 256, 256, 0, stream>>>(fc_w, fcwb, nw);

  mha_qkv_gemm<<<dim3(M_ / 64, E_ / 64, 3), 128, 0, stream>>>(
      xb, wqb, wkb, wvb, wq_b, wk_b, wv_b, Qb, Kb, Vtb);
  mha_scores<<<dim3(S_ / 64, B_ * H_), 128, 0, stream>>>(Qb, Kb, attn, dm);
  mha_softmax<<<dim3(B_ * H_ * S_), 256, 0, stream>>>(attn);
  mha_av<<<dim3(S_ / 64, B_ * H_), 128, 0, stream>>>(attn, Vtb, aob, dm);
  mha_fc<<<dim3(M_ / 64, E_ / 64), 128, 0, stream>>>(aob, fcwb, fc_b, out);

  (void)in_sizes; (void)n_in; (void)out_size; (void)ws_size;
}
